// Encoder_48000554500449
// MI455X (gfx1250) — matrix operand order repaired and checked by reading, not compiled, not run on hardware
//
#include <hip/hip_runtime.h>
#include <cstdint>
#include <cstddef>

#define T_LEN 512
#define BATCH 64
#define IDIM  256
#define HDIM  512
#define GDIM  (4 * HDIM)      // 2048 gate columns
#define NWG   32              // workgroups per direction (H-slices)
#define HSL   (HDIM / NWG)    // 16 h-columns per workgroup
#define NBLK  (GDIM / 16)     // 128 column blocks in swizzled weights

typedef __attribute__((ext_vector_type(16))) __bf16 v16bf;
typedef __attribute__((ext_vector_type(8)))  __bf16 v8bf;
typedef __attribute__((ext_vector_type(8)))  float  v8f;

union ABu { v16bf v; v8bf h[2]; };

// branchless activations (v_exp_f32 + fast div; no ocml branch sequences)
__device__ __forceinline__ float sigm(float x) {
    return __fdividef(1.0f, 1.0f + __expf(-x));
}
__device__ __forceinline__ float tanh_fast(float x) {
    return __fdividef(2.0f, 1.0f + __expf(-2.0f * x)) - 1.0f;
}

// ---------------- prep kernels ----------------

__global__ void k_zero_u32(unsigned* p, int n) {
    int i = blockIdx.x * blockDim.x + threadIdx.x;
    if (i < n) p[i] = 0u;
}

__global__ void k_f32_to_bf16(const float* __restrict__ src, __bf16* __restrict__ dst, long n) {
    long i = (long)blockIdx.x * blockDim.x + threadIdx.x;
    if (i < n) dst[i] = (__bf16)src[i];
}

// Swizzle W (GDIM x Ktot, row-major) into WMMA B-fragment order:
// block (kb,nb) covers K = kb*32..+31, N = nb*16..+15.
// lane<16: element j <-> (K=kb*32+j,     N=nb*16+lane)
// lane>=16: element j <-> (K=kb*32+16+j, N=nb*16+lane-16)
__global__ void k_swizzle_w(const float* __restrict__ W, __bf16* __restrict__ Wsw, int Ktot) {
    long e = (long)blockIdx.x * blockDim.x + threadIdx.x;
    long total = (long)Ktot * GDIM;
    if (e >= total) return;
    int j    = (int)(e & 15);
    int lane = (int)((e >> 4) & 31);
    long rest = e >> 9;
    int nb = (int)(rest & (NBLK - 1));
    int kb = (int)(rest >> 7);
    int k = kb * 32 + ((lane >> 4) << 4) + j;
    int n = nb * 16 + (lane & 15);
    Wsw[e] = (__bf16)W[(long)n * Ktot + k];
}

__global__ void k_bias_combine(const float* __restrict__ bi, const float* __restrict__ bh,
                               float* __restrict__ bc) {
    int g = blockIdx.x * blockDim.x + threadIdx.x;
    if (g < GDIM) bc[g] = bi[g] + bh[g];
}

// ---------------- scan kernel ----------------

__device__ __forceinline__ void grid_sync(unsigned* cnt, unsigned epoch) {
    __threadfence();
    __syncthreads();
    if (threadIdx.x == 0) {
        __hip_atomic_fetch_add(cnt, 1u, __ATOMIC_RELEASE, __HIP_MEMORY_SCOPE_AGENT);
        while (__hip_atomic_load(cnt, __ATOMIC_ACQUIRE, __HIP_MEMORY_SCOPE_AGENT) <
               (unsigned)NWG * epoch)
            __builtin_amdgcn_s_sleep(1);
    }
    __syncthreads();
}

// One K-segment of the gate GEMM: A rows stream from global (row-major bf16),
// B fragments stream from LDS-stationary swizzled weights.
// kb_lo/kb_hi/wofs are wave-uniform (SGPR) -> scalar loop control, EXEC all-1s.
__device__ __forceinline__ void mm_seg(const __bf16* __restrict__ arow,
                                       const __bf16* __restrict__ wlds,
                                       int kb_lo, int kb_hi, int wofs, int lane,
                                       v8f acc[4]) {
#pragma unroll 2
    for (int kb = kb_lo; kb < kb_hi; ++kb) {
        const __bf16* ap = arow + kb * 32;
        ABu a;
        a.h[0] = *(const v8bf*)(ap);
        a.h[1] = *(const v8bf*)(ap + 8);
#pragma unroll
        for (int g = 0; g < 4; ++g) {
            const __bf16* bp =
                wlds + ((((size_t)(wofs + kb) * 4 + g) * 32 + lane) << 4);
            ABu b;
            b.h[0] = *(const v8bf*)(bp);
            b.h[1] = *(const v8bf*)(bp + 8);
            acc[g] = __builtin_amdgcn_wmma_f32_16x16x32_bf16(
                false, a.v, false, b.v, (short)0, acc[g], false, false);
        }
    }
}

template <int IN_DIM, int LAYER>
__global__ void lstm_scan(const __bf16* __restrict__ xin,
                          const __bf16* __restrict__ WiF, const __bf16* __restrict__ WiR,
                          const __bf16* __restrict__ WhF, const __bf16* __restrict__ WhR,
                          const float* __restrict__ bF, const float* __restrict__ bR,
                          const float* __restrict__ h0, const float* __restrict__ c0,
                          __bf16* __restrict__ hping, float* __restrict__ cbuf,
                          __bf16* __restrict__ y0bf, float* __restrict__ out,
                          float* __restrict__ hn, float* __restrict__ cn,
                          unsigned* __restrict__ counters) {
    constexpr int KBX = IN_DIM / 32;        // x-part K blocks
    constexpr int KBH = HDIM / 32;          // h-part K blocks (16)
    constexpr int KBT = KBX + KBH;          // total K blocks
    constexpr int HALF = KBT / 2;           // K blocks per wave half

    extern __shared__ char smem[];
    __bf16* wlds = (__bf16*)smem;                         // [KBT][4][32][16] bf16 fragments
    float*  red  = (float*)(smem + (size_t)KBT * 4096);   // [4 rt][4 g][32 lane][8] f32

    const int dir  = blockIdx.y;
    const int bx   = blockIdx.x;
    const int s    = bx * HSL;              // this WG's h-column slice start
    const int lane = threadIdx.x & 31;
    // wave index is uniform across the wave: pin it to an SGPR so all loop
    // bounds / branches derived from it get scalar control flow (EXEC all-1s
    // around WMMA, as the ISA requires).
    const int wave = __builtin_amdgcn_readfirstlane(threadIdx.x >> 5);  // 0..7
    const int rt   = wave & 3;              // batch row tile 0..3
    const int kh   = wave >> 2;             // K half 0/1

    const __bf16* Wi = dir ? WiR : WiF;
    const __bf16* Wh = dir ? WhR : WhF;
    const float*  bs = dir ? bR : bF;
    unsigned* cnt = counters + LAYER * 2 + dir;
    __bf16* hp = hping + (size_t)dir * 2 * BATCH * HDIM;
    float*  cD = cbuf + (size_t)dir * BATCH * HDIM;
    const float* h0d = h0 + (size_t)(2 * LAYER + dir) * BATCH * HDIM;
    const float* c0d = c0 + (size_t)(2 * LAYER + dir) * BATCH * HDIM;

    // ---- prologue 1: stage this WG's weight fragments into LDS (reused 512x) ----
    for (int i = threadIdx.x; i < KBT * 4 * 32; i += blockDim.x) {
        const int l  = i & 31;
        const int g  = (i >> 5) & 3;
        const int kb = i >> 7;
        const int nbg = g * 32 + bx;        // global column block of this gate slice
        const __bf16* src = (kb < KBX)
            ? Wi + (((size_t)kb * NBLK + nbg) * 32 + l) * 16
            : Wh + (((size_t)(kb - KBX) * NBLK + nbg) * 32 + l) * 16;
        const uint4* s4 = (const uint4*)src;
        uint4* d4 = (uint4*)(wlds + ((size_t)i << 4));
        d4[0] = s4[0];
        d4[1] = s4[1];
    }
    // ---- prologue 2: stage initial h (bf16, ping buffer 0) and c (f32) slice ----
    for (int i = threadIdx.x; i < BATCH * HSL; i += blockDim.x) {
        int b = i / HSL, col = s + (i % HSL);
        hp[b * HDIM + col] = (__bf16)h0d[b * HDIM + col];
        cD[b * HDIM + col] = c0d[b * HDIM + col];
    }
    unsigned epoch = 1;
    grid_sync(cnt, epoch++);

    const int arow = rt * 16 + (lane & 15); // A-matrix row this lane supplies
    const int aoff = (lane >> 4) * 16;      // K offset of lane half (contiguous k = 16h + s, as B)
    const int kb_lo = kh * HALF;
    const int kb_hi = kb_lo + HALF;
    // this wave's share of the x-part and h-part K ranges (all wave-uniform)
    const int x_lo = (kb_lo < KBX) ? kb_lo : KBX;
    const int x_hi = (kb_hi < KBX) ? kb_hi : KBX;
    const int h_lo = (kb_lo > KBX) ? (kb_lo - KBX) : 0;
    const int h_hi = (kb_hi > KBX) ? (kb_hi - KBX) : 0;

    // t-invariant element-wise constants (used by kh==0 waves)
    const int colh = s + (lane & 15);       // C-layout: N = lane%16
    const float b0 = bs[0 * HDIM + colh];
    const float b1 = bs[1 * HDIM + colh];
    const float b2 = bs[2 * HDIM + colh];
    const float b3 = bs[3 * HDIM + colh];

    for (int t = 0; t < T_LEN; ++t) {
        const int tsrc = dir ? (T_LEN - 1 - t) : t;
        const __bf16* hread  = hp + (size_t)(t & 1) * BATCH * HDIM;
        __bf16*       hwrite = hp + (size_t)((t + 1) & 1) * BATCH * HDIM;

        const __bf16* xrow = xin + ((size_t)tsrc * BATCH + arow) * IN_DIM + aoff;
        const __bf16* hrow = hread + (size_t)arow * HDIM + aoff;

        v8f acc[4] = {};   // [gate] f32 accumulators, 16x16 tile each

        mm_seg(xrow, wlds, x_lo, x_hi, 0, lane, acc);     // x @ Wi^T slice
        mm_seg(hrow, wlds, h_lo, h_hi, KBX, lane, acc);   // h @ Wh^T slice

        // ---- cross-wave K reduction: upper half publishes, lower half combines ----
        if (kh == 1) {
#pragma unroll
            for (int g = 0; g < 4; ++g)
                *(v8f*)(red + ((size_t)(rt * 4 + g) * 32 + lane) * 8) = acc[g];
        }
        __syncthreads();

        if (kh == 0) {
            v8f tot[4];
#pragma unroll
            for (int g = 0; g < 4; ++g)
                tot[g] = acc[g] + *(const v8f*)(red + ((size_t)(rt * 4 + g) * 32 + lane) * 8);
#pragma unroll
            for (int j = 0; j < 8; ++j) {
                const int row = rt * 16 + j + (lane >> 4) * 8;   // C-layout: M
                const float ig = sigm(tot[0][j] + b0);
                const float fg = sigm(tot[1][j] + b1);
                const float gg = tanh_fast(tot[2][j] + b2);
                const float og = sigm(tot[3][j] + b3);
                const size_t cidx = (size_t)row * HDIM + colh;
                const float cnew = fg * cD[cidx] + ig * gg;
                const float hnew = og * tanh_fast(cnew);
                cD[cidx]     = cnew;
                hwrite[cidx] = (__bf16)hnew;
                const size_t yidx =
                    ((size_t)tsrc * BATCH + row) * (2 * HDIM) + (size_t)dir * HDIM + colh;
                if (LAYER == 0) y0bf[yidx] = (__bf16)hnew;
                else            out[yidx]  = hnew;
                if (t == T_LEN - 1) {
                    const size_t sidx = (size_t)(2 * LAYER + dir) * BATCH * HDIM + cidx;
                    hn[sidx] = hnew;
                    cn[sidx] = cnew;
                }
            }
        }
        grid_sync(cnt, epoch++);   // h ping-pong published for next step
    }
}

// ---------------- host launcher ----------------

extern "C" void kernel_launch(void* const* d_in, const int* in_sizes, int n_in,
                              void* d_out, int out_size, void* d_ws, size_t ws_size,
                              hipStream_t stream) {
    const float* x    = (const float*)d_in[0];
    const float* h0   = (const float*)d_in[1];
    const float* c0   = (const float*)d_in[2];
    const float* Wi0f = (const float*)d_in[3];
    const float* Wh0f = (const float*)d_in[4];
    const float* bi0f = (const float*)d_in[5];
    const float* bh0f = (const float*)d_in[6];
    const float* Wi0r = (const float*)d_in[7];
    const float* Wh0r = (const float*)d_in[8];
    const float* bi0r = (const float*)d_in[9];
    const float* bh0r = (const float*)d_in[10];
    const float* Wi1f = (const float*)d_in[11];
    const float* Wh1f = (const float*)d_in[12];
    const float* bi1f = (const float*)d_in[13];
    const float* bh1f = (const float*)d_in[14];
    const float* Wi1r = (const float*)d_in[15];
    const float* Wh1r = (const float*)d_in[16];
    const float* bi1r = (const float*)d_in[17];
    const float* bh1r = (const float*)d_in[18];

    float* out = (float*)d_out;
    float* hn  = out + (size_t)T_LEN * BATCH * 2 * HDIM;
    float* cn  = hn + (size_t)4 * BATCH * HDIM;

    // workspace carve-up (~103 MB)
    char* base = (char*)d_ws;
    size_t off = 0;
    auto take = [&](size_t bytes) -> char* {
        char* p = base + off;
        off = (off + bytes + 255) & ~(size_t)255;
        return p;
    };
    __bf16* xbf   = (__bf16*)take((size_t)T_LEN * BATCH * IDIM * 2);
    __bf16* y0bf  = (__bf16*)take((size_t)T_LEN * BATCH * 2 * HDIM * 2);
    __bf16* wi0fS = (__bf16*)take((size_t)GDIM * IDIM * 2);
    __bf16* wi0rS = (__bf16*)take((size_t)GDIM * IDIM * 2);
    __bf16* wh0fS = (__bf16*)take((size_t)GDIM * HDIM * 2);
    __bf16* wh0rS = (__bf16*)take((size_t)GDIM * HDIM * 2);
    __bf16* wi1fS = (__bf16*)take((size_t)GDIM * 2 * HDIM * 2);
    __bf16* wi1rS = (__bf16*)take((size_t)GDIM * 2 * HDIM * 2);
    __bf16* wh1fS = (__bf16*)take((size_t)GDIM * HDIM * 2);
    __bf16* wh1rS = (__bf16*)take((size_t)GDIM * HDIM * 2);
    float*  b0f   = (float*)take(GDIM * 4);
    float*  b0r   = (float*)take(GDIM * 4);
    float*  b1f   = (float*)take(GDIM * 4);
    float*  b1r   = (float*)take(GDIM * 4);
    __bf16* hping = (__bf16*)take((size_t)2 * 2 * BATCH * HDIM * 2); // dir x pingpong
    float*  cbuf  = (float*)take((size_t)2 * BATCH * HDIM * 4);
    unsigned* counters = (unsigned*)take(4 * sizeof(unsigned));

    k_zero_u32<<<1, 32, 0, stream>>>(counters, 4);

    {
        long n = (long)T_LEN * BATCH * IDIM;
        k_f32_to_bf16<<<(unsigned)((n + 255) / 256), 256, 0, stream>>>(x, xbf, n);
    }
    auto swz = [&](const float* W, __bf16* Wsw, int Ktot) {
        long n = (long)Ktot * GDIM;
        k_swizzle_w<<<(unsigned)((n + 255) / 256), 256, 0, stream>>>(W, Wsw, Ktot);
    };
    swz(Wi0f, wi0fS, IDIM);
    swz(Wi0r, wi0rS, IDIM);
    swz(Wh0f, wh0fS, HDIM);
    swz(Wh0r, wh0rS, HDIM);
    swz(Wi1f, wi1fS, 2 * HDIM);
    swz(Wi1r, wi1rS, 2 * HDIM);
    swz(Wh1f, wh1fS, HDIM);
    swz(Wh1r, wh1rS, HDIM);

    k_bias_combine<<<(GDIM + 255) / 256, 256, 0, stream>>>(bi0f, bh0f, b0f);
    k_bias_combine<<<(GDIM + 255) / 256, 256, 0, stream>>>(bi0r, bh0r, b0r);
    k_bias_combine<<<(GDIM + 255) / 256, 256, 0, stream>>>(bi1f, bh1f, b1f);
    k_bias_combine<<<(GDIM + 255) / 256, 256, 0, stream>>>(bi1r, bh1r, b1r);

    // dynamic LDS: weights-in-LDS needs > 64 KB per workgroup
    const int smem0 = ((IDIM + HDIM) / 32) * 4096 + 16384;      // 114,688 B
    const int smem1 = ((2 * HDIM + HDIM) / 32) * 4096 + 16384;  // 212,992 B
    (void)hipFuncSetAttribute(reinterpret_cast<const void*>(&lstm_scan<IDIM, 0>),
                              hipFuncAttributeMaxDynamicSharedMemorySize, smem0);
    (void)hipFuncSetAttribute(reinterpret_cast<const void*>(&lstm_scan<2 * HDIM, 1>),
                              hipFuncAttributeMaxDynamicSharedMemorySize, smem1);

    lstm_scan<IDIM, 0><<<dim3(NWG, 2), 256, smem0, stream>>>(
        xbf, wi0fS, wi0rS, wh0fS, wh0rS, b0f, b0r, h0, c0,
        hping, cbuf, y0bf, out, hn, cn, counters);

    lstm_scan<2 * HDIM, 1><<<dim3(NWG, 2), 256, smem1, stream>>>(
        y0bf, wi1fS, wi1rS, wh1fS, wh1rS, b1f, b1r, h0, c0,
        hping, cbuf, y0bf, out, hn, cn, counters);
}
